// MLP3_59751585022231
// MI455X (gfx1250) — matrix operand order repaired and checked by reading, not compiled, not run on hardware
//
#include <hip/hip_runtime.h>

// Problem constants (match reference setup)
constexpr int NSONGS = 100000;   // vocab size == padding id
constexpr int EMBED  = 64;
constexpr int BATCH  = 1024;
constexpr int SEQLEN = 200;

constexpr int NTPW            = 4;   // 16-column N-tiles per wave (64 columns/wave)
constexpr int WAVES_PER_BLOCK = 8;   // 256 threads / wave32

typedef __attribute__((ext_vector_type(16))) _Float16 v16h;
typedef __attribute__((ext_vector_type(8)))  float    v8f;

// ---------------------------------------------------------------------------
// Stage 1: masked mean pooling.  One 64-thread block per sample; thread t owns
// embedding dim t.  The song id is wave-uniform per position, so the validity
// branch is uniform.  Total traffic ~52 MB of 256B-row gathers -> ~3 us.
// ---------------------------------------------------------------------------
__global__ __launch_bounds__(EMBED) void pool_kernel(
    const int* __restrict__ songs, const float* __restrict__ emb,
    float* __restrict__ pooled) {
  const int b = blockIdx.x;
  const int t = threadIdx.x;
  const int* row = songs + (size_t)b * SEQLEN;
  float acc = 0.0f;
  int count = 0;
  for (int l = 0; l < SEQLEN; ++l) {
    const int id = row[l];                 // uniform across the block
    if (id != NSONGS) {
      ++count;
      acc += emb[(size_t)id * EMBED + t];
    }
  }
  const float cnt = (count > 0) ? (float)count : 1.0f;   // lengths >= 1
  pooled[(size_t)b * EMBED + t] = acc / cnt;
}

// ---------------------------------------------------------------------------
// Stage 2: [1024,64] x [64,100000] GEMM + bias + sigmoid via
// v_wmma_f32_16x16x32_f16.  K=64 -> 2 WMMAs per 16x16 output tile.
// Each wave: 4 N-tiles resident in registers, loop over 64 M-tiles.
// ---------------------------------------------------------------------------

// A-matrix 16x32 f16 layout (matches B): lanes 0-15 hold row M=lane with
// halves[0..15]=K ka..ka+15 (ka=0); lanes 16-31 hold the same row with
// ka=16.
__device__ inline v16h load_cvt_A(const float* __restrict__ arow, int ka) {
  v16h a;
#pragma unroll
  for (int i = 0; i < 8; ++i) {
    a[i]     = (_Float16)arow[ka + i];
    a[8 + i] = (_Float16)arow[ka + 8 + i];
  }
  return a;
}

// B-matrix 32x16 f16 layout: lane = column N (mod 16); lanes 0-15 hold
// K kb..kb+15 (kb=0), lanes 16-31 hold kb=16 -> 16 contiguous floats per lane.
__device__ inline v16h load_cvt_B(const float* __restrict__ wrow, int kb) {
  v16h b;
#pragma unroll
  for (int i = 0; i < 16; ++i) b[i] = (_Float16)wrow[kb + i];
  return b;
}

__global__ __launch_bounds__(WAVES_PER_BLOCK * 32) void gemm_sigmoid_kernel(
    const float* __restrict__ pooled,   // [BATCH, EMBED]
    const float* __restrict__ W,        // [NSONGS, EMBED] row-major
    const float* __restrict__ bias,     // [NSONGS]
    float* __restrict__ out) {          // [BATCH, NSONGS]
  const int lane      = threadIdx.x & 31;
  const int waveInBlk = threadIdx.x >> 5;
  const int waveId    = blockIdx.x * WAVES_PER_BLOCK + waveInBlk;

  const int tileBase = waveId * NTPW;          // first 16-col tile index
  const int nTiles   = NSONGS / 16;            // 6250 (exact)
  int jcount = nTiles - tileBase;              // wave-uniform
  if (jcount <= 0) return;                     // whole wave exits -> EXEC all-1s below
  if (jcount > NTPW) jcount = NTPW;

  const int  colInTile = lane & 15;
  const bool hiHalf    = lane >= 16;
  const int  ka        = hiHalf ? 16 : 0;
  const int  kb        = hiHalf ? 16 : 0;

  // Resident B tiles (W slice, f32 -> f16) + per-lane bias.
  v16h blo[NTPW], bhi[NTPW];
  float bvals[NTPW];
#pragma unroll
  for (int j = 0; j < NTPW; ++j) {
    if (j < jcount) {
      const int n = (tileBase + j) * 16 + colInTile;
      const float* wrow = W + (size_t)n * EMBED;
      blo[j]   = load_cvt_B(wrow, kb);         // K  0..31 slice
      bhi[j]   = load_cvt_B(wrow, kb + 32);    // K 32..63 slice
      bvals[j] = bias[n];
    }
  }

  constexpr float NEG_LOG2E = -1.44269504088896340736f;

  for (int mt = 0; mt < BATCH / 16; ++mt) {
    const float* arow = pooled + (size_t)(mt * 16 + colInTile) * EMBED;
    const v16h alo = load_cvt_A(arow, ka);        // K  0..31
    const v16h ahi = load_cvt_A(arow, 32 + ka);   // K 32..63

#pragma unroll
    for (int j = 0; j < NTPW; ++j) {
      if (j >= jcount) break;                     // wave-uniform
      v8f c = {};
      c = __builtin_amdgcn_wmma_f32_16x16x32_f16(false, alo, false, blo[j],
                                                 (short)0, c, false, false);
      c = __builtin_amdgcn_wmma_f32_16x16x32_f16(false, ahi, false, bhi[j],
                                                 (short)0, c, false, false);

      // C/D layout: element r -> row M = mt*16 + r + (hiHalf?8:0), col N below.
      const int    n     = (tileBase + j) * 16 + colInTile;
      const int    rbase = mt * 16 + (hiHalf ? 8 : 0);
      const float  bj    = bvals[j];
#pragma unroll
      for (int r = 0; r < 8; ++r) {
        const float x = c[r] + bj;
        const float e = __builtin_amdgcn_exp2f(x * NEG_LOG2E);  // exp(-x)
        const float s = __builtin_amdgcn_rcpf(1.0f + e);        // sigmoid
        out[(size_t)(rbase + r) * NSONGS + n] = s;
      }
    }
  }
}

// ---------------------------------------------------------------------------
// d_in order: songs[int, B*L], emb_table[f32, NSONGS*64], W[f32, NSONGS*64],
// b[f32, NSONGS].  d_out: f32 [B, NSONGS].  d_ws: pooled [B, 64] f32 (256 KB).
// ---------------------------------------------------------------------------
extern "C" void kernel_launch(void* const* d_in, const int* in_sizes, int n_in,
                              void* d_out, int out_size, void* d_ws, size_t ws_size,
                              hipStream_t stream) {
  const int*   songs = (const int*)d_in[0];
  const float* emb   = (const float*)d_in[1];
  const float* W     = (const float*)d_in[2];
  const float* bias  = (const float*)d_in[3];
  float*       out   = (float*)d_out;
  float*       pooled = (float*)d_ws;   // BATCH*EMBED*4 = 256 KB scratch

  pool_kernel<<<BATCH, EMBED, 0, stream>>>(songs, emb, pooled);

  const int nTiles  = NSONGS / 16;                               // 6250
  const int nWaves  = (nTiles + NTPW - 1) / NTPW;                // 1563
  const int nBlocks = (nWaves + WAVES_PER_BLOCK - 1) / WAVES_PER_BLOCK;  // 196
  gemm_sigmoid_kernel<<<nBlocks, WAVES_PER_BLOCK * 32, 0, stream>>>(
      pooled, W, bias, out);
}
